// MultiHeadAttention_5592047419650
// MI455X (gfx1250) — matrix operand order repaired and checked by reading, not compiled, not run on hardware
//
#include <hip/hip_runtime.h>

// Problem dimensions
constexpr int B_  = 128;
constexpr int T_  = 256;
constexpr int C_  = 384;
constexpr int H_  = 6;
constexpr int HS_ = 64;
constexpr int NC_ = C_ / 32;                     // 12 K-chunks of 32

typedef __attribute__((ext_vector_type(16))) _Float16 v16h;
typedef __attribute__((ext_vector_type(8)))  _Float16 v8h;
typedef __attribute__((ext_vector_type(4)))  _Float16 v4h;
typedef __attribute__((ext_vector_type(8)))  float    v8f;

// ---------------------------------------------------------------------------
// WMMA helpers (CDNA5 16x16x32 f16 -> f32)
// ---------------------------------------------------------------------------
__device__ __forceinline__ v8f wmma_f16(v16h a, v16h b, v8f c) {
    // (neg_a, A, neg_b, B, c_mod, C, reuse_a, reuse_b)
    return __builtin_amdgcn_wmma_f32_16x16x32_f16(false, a, false, b, (short)0, c,
                                                  false, false);
}

__device__ __forceinline__ v16h combine8(v8h lo, v8h hi) {
    return __builtin_shufflevector(lo, hi, 0, 1, 2, 3, 4, 5, 6, 7,
                                           8, 9, 10, 11, 12, 13, 14, 15);
}

// A-fragment (16x32 f16, M x K) from a row-major row pointer already offset to
// the K-chunk base.  lhi = lane>>4.
// half h of lane L holds K = lhi*16 + h  ->  two contiguous
// 8-half (16B) loads at +lhi*16 and +lhi*16+8.
__device__ __forceinline__ v16h load_a_frag(const _Float16* __restrict__ row, int lhi) {
    v8h lo = *(const v8h*)(row + lhi * 16);
    v8h hi = *(const v8h*)(row + lhi * 16 + 8);
    return combine8(lo, hi);
}

// Row reductions within a 16-lane half-wave (D-layout: each 16x16 row lives in
// one half of the wave).
__device__ __forceinline__ float rowmax16(float v) {
    v = fmaxf(v, __shfl_xor(v, 1, 32));
    v = fmaxf(v, __shfl_xor(v, 2, 32));
    v = fmaxf(v, __shfl_xor(v, 4, 32));
    v = fmaxf(v, __shfl_xor(v, 8, 32));
    return v;
}
__device__ __forceinline__ float rowsum16(float v) {
    v += __shfl_xor(v, 1, 32);
    v += __shfl_xor(v, 2, 32);
    v += __shfl_xor(v, 4, 32);
    v += __shfl_xor(v, 8, 32);
    return v;
}

// Prefetch a 16-row x 384-half (768B) f16 tile: 96 cachelines, 3 instructions.
// Lane L touches row llo, line (p*2 + lhi) of 6.
__device__ __forceinline__ void prefetch_a_tile(const _Float16* base, int lhi, int llo) {
    const _Float16* row = base + (size_t)llo * C_;
#pragma unroll
    for (int p = 0; p < 3; ++p)
        __builtin_prefetch(row + (p * 2 + lhi) * 64, 0, 0);
}

// ---------------------------------------------------------------------------
// Prepack kernels
// ---------------------------------------------------------------------------
__global__ __launch_bounds__(256) void cast_x_kernel(const float* __restrict__ x,
                                                     _Float16* __restrict__ xh,
                                                     int n4) {
    int i = blockIdx.x * 256 + threadIdx.x;
    if (i < n4) {
        float4 v = ((const float4*)x)[i];
        v4h o;
        o[0] = (_Float16)v.x; o[1] = (_Float16)v.y;
        o[2] = (_Float16)v.z; o[3] = (_Float16)v.w;
        ((v4h*)xh)[i] = o;
    }
}

// W[h][c][d] (f32) -> wT[h][d][c] (f16): B-fragment order for the QKV GEMMs.
__global__ __launch_bounds__(256) void pack_w_kernel(const float* __restrict__ Wq,
                                                     const float* __restrict__ Wk,
                                                     const float* __restrict__ Wv,
                                                     _Float16* __restrict__ wqT,
                                                     _Float16* __restrict__ wkT,
                                                     _Float16* __restrict__ wvT) {
    int i = blockIdx.x * 256 + threadIdx.x;          // over H*HS*C
    if (i >= H_ * HS_ * C_) return;
    int c  = i % C_;
    int hd = i / C_;
    int d  = hd % HS_;
    int h  = hd / HS_;
    size_t src = ((size_t)h * C_ + c) * HS_ + d;
    wqT[i] = (_Float16)Wq[src];
    wkT[i] = (_Float16)Wk[src];
    wvT[i] = (_Float16)Wv[src];
}

__global__ __launch_bounds__(256) void cast_wp_kernel(const float* __restrict__ Wp,
                                                      _Float16* __restrict__ wph) {
    int i = blockIdx.x * 256 + threadIdx.x;
    if (i < C_ * C_) wph[i] = (_Float16)Wp[i];       // Wp[n][c] is already B-frag order
}

// ---------------------------------------------------------------------------
// Kernel 1: QKV projection.  One block per (b,h); each wave owns a fixed
// d-tile (tile & 3 invariant), so all 36 weight B-fragments stay register-
// resident across the t-loop.  A rows 128 t ahead are prefetched
// (global_prefetch_b8, no LOADcnt) while the current tile computes, so the
// scheduler's load->wait->wmma pattern hits warm cache.
// Q,K stored row-major [t][d]; V stored transposed [d][t].
// ---------------------------------------------------------------------------
__global__ __launch_bounds__(256) void qkv_kernel(const _Float16* __restrict__ xh,
                                                  const _Float16* __restrict__ wqT,
                                                  const _Float16* __restrict__ wkT,
                                                  const _Float16* __restrict__ wvT,
                                                  _Float16* __restrict__ Qh,
                                                  _Float16* __restrict__ Kh,
                                                  _Float16* __restrict__ Vt) {
    const int bh   = blockIdx.x;                 // b*H + h
    const int b    = bh / H_;
    const int h    = bh % H_;
    const int wave = threadIdx.x >> 5;
    const int lane = threadIdx.x & 31;
    const int lhi  = lane >> 4;
    const int llo  = lane & 15;

    const _Float16* xrow = xh  + (size_t)b * T_ * C_;
    const _Float16* wq   = wqT + (size_t)h * HS_ * C_;
    const _Float16* wk   = wkT + (size_t)h * HS_ * C_;
    const _Float16* wv   = wvT + (size_t)h * HS_ * C_;
    _Float16* Qo = Qh + (size_t)bh * T_ * HS_;
    _Float16* Ko = Kh + (size_t)bh * T_ * HS_;
    _Float16* Vo = Vt + (size_t)bh * HS_ * T_;

    // warm the first tile before entering the loop
    prefetch_a_tile(xrow + (size_t)((wave >> 2) * 16) * C_, lhi, llo);

    // 16 (t-tiles) x 4 (d-tiles) = 64 tiles; wave keeps d0 = (wave&3)*16 fixed
    for (int tile = wave; tile < 64; tile += 8) {
        const int d0 = (tile & 3) * 16;
        const int t0 = (tile >> 2) * 16;
        const _Float16* arow = xrow + (size_t)(t0 + llo) * C_;

        // prefetch the A rows 128 ahead (t0 advances by 32 per iteration)
        if (tile + 8 < 64)
            prefetch_a_tile(xrow + (size_t)(t0 + 128) * C_, lhi, llo);

        v8f accq = {}, acck = {}, accv = {};
#pragma unroll
        for (int ct = 0; ct < NC_; ++ct) {
            v16h a = load_a_frag(arow + ct * 32, lhi);
            const size_t boff = (size_t)(d0 + llo) * C_ + ct * 32 + lhi * 16;
            accq = wmma_f16(a, *(const v16h*)(wq + boff), accq);
            acck = wmma_f16(a, *(const v16h*)(wk + boff), acck);
            accv = wmma_f16(a, *(const v16h*)(wv + boff), accv);
        }
        // Q, K row-major: lane has fixed d, rows vary -> scalar f16 stores
#pragma unroll
        for (int r = 0; r < 8; ++r) {
            const int t = t0 + r + lhi * 8;
            const int d = d0 + llo;
            Qo[(size_t)t * HS_ + d] = (_Float16)accq[r];
            Ko[(size_t)t * HS_ + d] = (_Float16)acck[r];
        }
        // V transposed: lane has fixed d, 8 consecutive t -> one 16B store
        v8h vv;
#pragma unroll
        for (int r = 0; r < 8; ++r) vv[r] = (_Float16)accv[r];
        *(v8h*)(Vo + (size_t)(d0 + llo) * T_ + t0 + lhi * 8) = vv;
    }
}

// ---------------------------------------------------------------------------
// Kernel 2: flash attention per (b,h).  Wave w processes row-tiles {w, 15-w}
// (exactly 9 s-chunks each).  V fragments are loaded at the top of each
// s-chunk (consumed after softmax); next chunk's K and V lines are prefetched
// during the current chunk.  P converts D->A layout via per-wave LDS.
// ---------------------------------------------------------------------------
__global__ __launch_bounds__(256) void attn_kernel(const _Float16* __restrict__ Qh,
                                                   const _Float16* __restrict__ Kh,
                                                   const _Float16* __restrict__ Vt,
                                                   _Float16* __restrict__ att) {
    __shared__ _Float16 pbuf[8][16][32];         // per-wave P staging (8 KB)

    const int bh   = blockIdx.x;
    const int b    = bh / H_;
    const int h    = bh % H_;
    const int wave = threadIdx.x >> 5;
    const int lane = threadIdx.x & 31;
    const int lhi  = lane >> 4;
    const int llo  = lane & 15;

    const _Float16* Q = Qh + (size_t)bh * T_ * HS_;
    const _Float16* K = Kh + (size_t)bh * T_ * HS_;
    const _Float16* V = Vt + (size_t)bh * HS_ * T_;
    _Float16* O = att + (size_t)b * T_ * C_ + h * HS_;

    const float scale = 0.125f;                  // 1/sqrt(64)
    const float NEG   = -3.0e38f;

    const int its[2] = { wave, 15 - wave };      // balanced causal pairing
#pragma unroll
    for (int p = 0; p < 2; ++p) {
        const int t0 = its[p] * 16;
        // Q A-fragments for both 32-wide K-chunks of HS=64
        v16h qa0 = load_a_frag(Q + (size_t)(t0 + llo) * HS_ + 0,  lhi);
        v16h qa1 = load_a_frag(Q + (size_t)(t0 + llo) * HS_ + 32, lhi);

        v8f o0 = {}, o1 = {}, o2 = {}, o3 = {};
        float m[8], l[8];
#pragma unroll
        for (int r = 0; r < 8; ++r) { m[r] = NEG; l[r] = 0.0f; }

        const int jmax = (t0 + 15) >> 5;         // last 32-wide s-chunk

        for (int jj = 0; jj <= jmax; ++jj) {
            const int s0 = jj * 32;

            if (jj < jmax) {
                // prefetch next chunk's K rows (32 rows x 1 line each)
                __builtin_prefetch(K + (size_t)(s0 + 32 + lane) * HS_, 0, 0);
                // prefetch next chunk's V lines (64 d-rows x 1 line each)
                __builtin_prefetch(V + (size_t)(lane) * T_ + s0 + 32, 0, 0);
                __builtin_prefetch(V + (size_t)(lane + 32) * T_ + s0 + 32, 0, 0);
            }

            // ---- V B-fragments: in flight during scores + softmax ----
            const _Float16* vb = V + (size_t)llo * T_ + s0 + lhi * 16;
            v16h vf0 = *(const v16h*)(vb + (size_t)0  * T_);
            v16h vf1 = *(const v16h*)(vb + (size_t)16 * T_);
            v16h vf2 = *(const v16h*)(vb + (size_t)32 * T_);
            v16h vf3 = *(const v16h*)(vb + (size_t)48 * T_);

            // ---- scores: two 16x16 D tiles over a 32-wide s-chunk ----
            v8f sA = {}, sB = {};
            {
                const _Float16* k0 = K + (size_t)(s0 + llo) * HS_ + lhi * 16;
                const _Float16* k1 = K + (size_t)(s0 + 16 + llo) * HS_ + lhi * 16;
                v16h b00 = *(const v16h*)(k0);
                v16h b01 = *(const v16h*)(k0 + 32);
                v16h b10 = *(const v16h*)(k1);
                v16h b11 = *(const v16h*)(k1 + 32);
                sA = wmma_f16(qa0, b00, sA);
                sA = wmma_f16(qa1, b01, sA);
                sB = wmma_f16(qa0, b10, sB);
                sB = wmma_f16(qa1, b11, sB);
            }

            // ---- scale + causal mask ----
            const bool need_mask = (s0 + 31 > t0);
            float scA[8], scB[8];
#pragma unroll
            for (int r = 0; r < 8; ++r) {
                const int trow = t0 + r + lhi * 8;
                float a  = sA[r] * scale;
                float bb = sB[r] * scale;
                if (need_mask) {
                    if (s0 + llo > trow)      a  = NEG;
                    if (s0 + 16 + llo > trow) bb = NEG;
                }
                scA[r] = a; scB[r] = bb;
            }

            // ---- online softmax update ----
            float alpha[8];
#pragma unroll
            for (int r = 0; r < 8; ++r) {
                const float rm = rowmax16(fmaxf(scA[r], scB[r]));
                const float mn = fmaxf(m[r], rm);
                alpha[r] = __expf(m[r] - mn);
                const float pa = __expf(scA[r] - mn);
                const float pb = __expf(scB[r] - mn);
                scA[r] = pa; scB[r] = pb;
                l[r] = l[r] * alpha[r] + rowsum16(pa + pb);
                m[r] = mn;
            }
#pragma unroll
            for (int r = 0; r < 8; ++r) {
                o0[r] *= alpha[r]; o1[r] *= alpha[r];
                o2[r] *= alpha[r]; o3[r] *= alpha[r];
            }

            // ---- P: D-layout -> A-layout via per-wave LDS tile ----
#pragma unroll
            for (int r = 0; r < 8; ++r) {
                const int mr = r + lhi * 8;
                pbuf[wave][mr][llo]      = (_Float16)scA[r];
                pbuf[wave][mr][16 + llo] = (_Float16)scB[r];
            }
            v16h pa = load_a_frag(&pbuf[wave][llo][0], lhi);

            // ---- O += P x V ----
            o0 = wmma_f16(pa, vf0, o0);
            o1 = wmma_f16(pa, vf1, o1);
            o2 = wmma_f16(pa, vf2, o2);
            o3 = wmma_f16(pa, vf3, o3);
        }

        // ---- normalize and store f16 att rows ----
#pragma unroll
        for (int r = 0; r < 8; ++r) {
            const float inv = 1.0f / l[r];
            const int t = t0 + r + lhi * 8;
            _Float16* orow = O + (size_t)t * C_;
            orow[0  + llo] = (_Float16)(o0[r] * inv);
            orow[16 + llo] = (_Float16)(o1[r] * inv);
            orow[32 + llo] = (_Float16)(o2[r] * inv);
            orow[48 + llo] = (_Float16)(o3[r] * inv);
        }
    }
}

// ---------------------------------------------------------------------------
// Kernel 3: output projection  out = att @ Wp.T + bp  (f32 out)
// Each wave fixes one n-tile, keeps the 12 Wp B-fragments register-resident,
// loops over 4 m-tiles, prefetching the next m-tile's att rows.
// Grid: (24 n-tiles, 64 m-groups of 32 tiles).
// ---------------------------------------------------------------------------
__global__ __launch_bounds__(256) void proj_kernel(const _Float16* __restrict__ att,
                                                   const _Float16* __restrict__ wph,
                                                   const float* __restrict__ bp,
                                                   float* __restrict__ out) {
    const int wave = threadIdx.x >> 5;
    const int lane = threadIdx.x & 31;
    const int lhi  = lane >> 4;
    const int llo  = lane & 15;

    const int n0    = blockIdx.x * 16;           // fixed per block
    const int mbase = blockIdx.y * 32;           // 32 m-tiles per block

    // Wp B-fragments: register-resident across all 4 m-tiles
    v16h bfr[NC_];
#pragma unroll
    for (int ct = 0; ct < NC_; ++ct)
        bfr[ct] = *(const v16h*)(wph + (size_t)(n0 + llo) * C_ + ct * 32 + lhi * 16);
    const float bias = bp[n0 + llo];

    // warm the first m-tile
    prefetch_a_tile(att + (size_t)((mbase + wave) * 16) * C_, lhi, llo);

#pragma unroll
    for (int k = 0; k < 4; ++k) {
        const int m0 = (mbase + k * 8 + wave) * 16;
        const _Float16* arow = att + (size_t)(m0 + llo) * C_;

        if (k + 1 < 4)                           // next m-tile is +128 rows
            prefetch_a_tile(att + (size_t)(m0 + 128) * C_, lhi, llo);

        v8f acc = {};
#pragma unroll
        for (int ct = 0; ct < NC_; ++ct) {
            v16h a = load_a_frag(arow + ct * 32, lhi);
            acc = wmma_f16(a, bfr[ct], acc);
        }
#pragma unroll
        for (int r = 0; r < 8; ++r)
            out[(size_t)(m0 + r + lhi * 8) * C_ + n0 + llo] = acc[r] + bias;
    }
}

// ---------------------------------------------------------------------------
// Host-side launcher
// ---------------------------------------------------------------------------
extern "C" void kernel_launch(void* const* d_in, const int* in_sizes, int n_in,
                              void* d_out, int out_size, void* d_ws, size_t ws_size,
                              hipStream_t stream) {
    (void)in_sizes; (void)n_in; (void)out_size; (void)ws_size;

    const float* x  = (const float*)d_in[0];
    const float* Wq = (const float*)d_in[1];
    const float* Wk = (const float*)d_in[2];
    const float* Wv = (const float*)d_in[3];
    const float* Wp = (const float*)d_in[4];
    const float* bp = (const float*)d_in[5];
    float* out = (float*)d_out;

    // Workspace carve-out (all f16), 256B aligned
    char* ws = (char*)d_ws;
    size_t off = 0;
    auto carve = [&](size_t bytes) -> char* {
        char* p = ws + off;
        off = (off + bytes + 255) & ~(size_t)255;
        return p;
    };
    const size_t xh_b  = (size_t)B_ * T_ * C_ * 2;
    const size_t wt_b  = (size_t)H_ * HS_ * C_ * 2;
    const size_t wp_b  = (size_t)C_ * C_ * 2;
    const size_t qkv_b = (size_t)B_ * H_ * T_ * HS_ * 2;
    const size_t att_b = (size_t)B_ * T_ * C_ * 2;

    _Float16* xh   = (_Float16*)carve(xh_b);
    _Float16* wqT  = (_Float16*)carve(wt_b);
    _Float16* wkT  = (_Float16*)carve(wt_b);
    _Float16* wvT  = (_Float16*)carve(wt_b);
    _Float16* wph  = (_Float16*)carve(wp_b);
    _Float16* Qhp  = (_Float16*)carve(qkv_b);
    _Float16* Khp  = (_Float16*)carve(qkv_b);
    _Float16* Vtp  = (_Float16*)carve(qkv_b);
    _Float16* attb = (_Float16*)carve(att_b);

    // 1) prepack
    const int n4 = B_ * T_ * C_ / 4;                       // 3,145,728
    cast_x_kernel<<<n4 / 256, 256, 0, stream>>>(x, xh, n4);
    pack_w_kernel<<<(H_ * HS_ * C_) / 256, 256, 0, stream>>>(Wq, Wk, Wv, wqT, wkT, wvT);
    cast_wp_kernel<<<(C_ * C_) / 256, 256, 0, stream>>>(Wp, wph);

    // 2) QKV projection: one block per (b,h)
    qkv_kernel<<<B_ * H_, 256, 0, stream>>>(xh, wqT, wkT, wvT, Qhp, Khp, Vtp);

    // 3) causal flash attention: one block per (b,h)
    attn_kernel<<<B_ * H_, 256, 0, stream>>>(Qhp, Khp, Vtp, attb);

    // 4) output projection
    proj_kernel<<<dim3(C_ / 16, 64), 256, 0, stream>>>(attb, wph, bp, out);
}
